// TopKMoEClassifier_17660905521548
// MI455X (gfx1250) — matrix operand order repaired and checked by reading, not compiled, not run on hardware
//
#include <hip/hip_runtime.h>
#include <hip/hip_bf16.h>
#include <cstdint>
#include <cstddef>

// ---------------- problem sizes ----------------
constexpr int Bsz = 8192;   // tokens
constexpr int Dsz = 4096;   // hidden
constexpr int Esz = 16;     // experts
constexpr int Osz = 512;    // output dim
constexpr int TOPK = 2;

typedef __attribute__((ext_vector_type(16))) __bf16 v16bf;
typedef __attribute__((ext_vector_type(8)))  float  v8f;

struct U16x16 { uint4 lo, hi; };

__device__ inline v16bf make_frag(uint4 lo, uint4 hi) {
    U16x16 t{lo, hi};
    return __builtin_bit_cast(v16bf, t);
}

// round-to-nearest-even f32 -> bf16 (bits)
__device__ inline unsigned short f2bf(float f) {
    unsigned u = __builtin_bit_cast(unsigned, f);
    u += 0x7FFFu + ((u >> 16) & 1u);
    return (unsigned short)(u >> 16);
}

// ---- CDNA5 async global->LDS copy (ASYNCcnt-tracked DMA into LDS) ----
// VDST = LDS byte offset (wave-relative, i.e. low 32 bits of a __shared__
// generic pointer), VADDR = 64-bit global address, GV mode (saddr = off).
__device__ inline void async_ld_b128(unsigned ldsOff, const void* g) {
    asm volatile("global_load_async_to_lds_b128 %0, %1, off"
                 :: "v"(ldsOff), "v"(g) : "memory");
}
__device__ inline void wait_async0() {
#if __has_builtin(__builtin_amdgcn_s_wait_asynccnt)
    __builtin_amdgcn_s_wait_asynccnt(0);
#else
    asm volatile("s_wait_asynccnt 0x0" ::: "memory");
#endif
}

// ---------------- 0: init output & counters ----------------
__global__ void k_init(float* __restrict__ out, int n,
                       float* __restrict__ pi_sum,
                       unsigned* __restrict__ counts,
                       unsigned* __restrict__ cursors) {
    int i = blockIdx.x * blockDim.x + threadIdx.x;
    if (i < n) out[i] = 0.0f;
    if (i < Esz) { pi_sum[i] = 0.0f; counts[i] = 0u; cursors[i] = 0u; }
}

// ---------------- 1: fp32 -> bf16, same layout (features) ----------------
__global__ void k_convert_rows(const float* __restrict__ src,
                               unsigned short* __restrict__ dst, int n8) {
    int i = blockIdx.x * blockDim.x + threadIdx.x;
    if (i >= n8) return;
    const float4* s = (const float4*)src + (size_t)i * 2;
    float4 a = s[0], b = s[1];
    uint4 v;
    v.x = (unsigned)f2bf(a.x) | ((unsigned)f2bf(a.y) << 16);
    v.y = (unsigned)f2bf(a.z) | ((unsigned)f2bf(a.w) << 16);
    v.z = (unsigned)f2bf(b.x) | ((unsigned)f2bf(b.y) << 16);
    v.w = (unsigned)f2bf(b.z) | ((unsigned)f2bf(b.w) << 16);
    ((uint4*)dst)[i] = v;
}

// ---------------- 2: W_e [E][D][O] fp32 -> W_eT [E][O][D] bf16 ----------------
__global__ void k_convert_weT(const float* __restrict__ We,
                              unsigned short* __restrict__ WeT) {
    int gid = blockIdx.x * blockDim.x + threadIdx.x;   // E*O*(D/4)
    int o = gid % Osz;
    int t = gid / Osz;
    int d = (t % (Dsz / 4)) * 4;
    int e = t / (Dsz / 4);
    if (e >= Esz) return;
    size_t s = ((size_t)e * Dsz + d) * Osz + o;
    unsigned short r0 = f2bf(We[s]);
    unsigned short r1 = f2bf(We[s + Osz]);
    unsigned short r2 = f2bf(We[s + 2 * (size_t)Osz]);
    unsigned short r3 = f2bf(We[s + 3 * (size_t)Osz]);
    uint2 v;
    v.x = (unsigned)r0 | ((unsigned)r1 << 16);
    v.y = (unsigned)r2 | ((unsigned)r3 << 16);
    *(uint2*)(WeT + ((size_t)e * Osz + o) * Dsz + d) = v;
}

// ---------------- 3: router (one wave32 per token) ----------------
__global__ __launch_bounds__(256)
void k_router(const float* __restrict__ feat, const float* __restrict__ Wr,
              const float* __restrict__ br,
              int* __restrict__ top_i, float* __restrict__ top_w,
              unsigned* __restrict__ counts, float* __restrict__ blockProb) {
    __shared__ float sP[8][Esz];
    int wid  = threadIdx.x >> 5;
    int lane = threadIdx.x & 31;
    int tok  = blockIdx.x * 8 + wid;

    const float* f = feat + (size_t)tok * Dsz;
    float acc[Esz];
#pragma unroll
    for (int e = 0; e < Esz; ++e) acc[e] = 0.0f;

    for (int d = lane; d < Dsz; d += 32) {
        float fv = f[d];
        const float* wr = Wr + (size_t)d * Esz;
#pragma unroll
        for (int e = 0; e < Esz; ++e) acc[e] += fv * wr[e];
    }
    // butterfly reduce each expert logit across the wave
#pragma unroll
    for (int e = 0; e < Esz; ++e) {
        float v = acc[e];
#pragma unroll
        for (int off = 16; off >= 1; off >>= 1) v += __shfl_xor(v, off, 32);
        acc[e] = v;
    }

    if (lane == 0) {
        float mx = -3.4e38f;
#pragma unroll
        for (int e = 0; e < Esz; ++e) { acc[e] += br[e]; mx = fmaxf(mx, acc[e]); }
        float s = 0.0f;
        float p[Esz];
#pragma unroll
        for (int e = 0; e < Esz; ++e) { p[e] = expf(acc[e] - mx); s += p[e]; }
        float inv = 1.0f / s;
#pragma unroll
        for (int e = 0; e < Esz; ++e) { p[e] *= inv; sP[wid][e] = p[e]; }
        // top-2 (first index wins ties, matching top_k)
        int i0 = 0;
#pragma unroll
        for (int e = 1; e < Esz; ++e) if (p[e] > p[i0]) i0 = e;
        int i1 = (i0 == 0) ? 1 : 0;
#pragma unroll
        for (int e = 0; e < Esz; ++e) if (e != i0 && p[e] > p[i1]) i1 = e;
        float v0 = p[i0], v1 = p[i1];
        float s2 = fmaxf(v0 + v1, 1e-9f);
        top_i[tok * 2 + 0] = i0;  top_w[tok * 2 + 0] = v0 / s2;
        top_i[tok * 2 + 1] = i1;  top_w[tok * 2 + 1] = v1 / s2;
        atomicAdd(&counts[i0], 1u);
        atomicAdd(&counts[i1], 1u);
    }
    __syncthreads();
    // deterministic per-block pi partial sum
    if (threadIdx.x < Esz) {
        float s = 0.0f;
#pragma unroll
        for (int w = 0; w < 8; ++w) s += sP[w][threadIdx.x];
        blockProb[blockIdx.x * Esz + threadIdx.x] = s;
    }
}

// ---------------- 4: exclusive scan + balance loss ----------------
__global__ void k_finalize(const unsigned* __restrict__ counts,
                           unsigned* __restrict__ offsets,
                           const float* __restrict__ blockProb, int nblocks,
                           float* __restrict__ loss_out) {
    __shared__ float term[Esz];
    int t = threadIdx.x;
    if (t < Esz) {
        float s = 0.0f;
        for (int b = 0; b < nblocks; ++b) s += blockProb[b * Esz + t];
        float pe = s / (float)Bsz;
        term[t] = pe * logf(fmaxf(pe, 1e-9f));
    }
    __syncthreads();
    if (t == 0) {
        float l = 0.0f;
#pragma unroll
        for (int e = 0; e < Esz; ++e) l += term[e];
        loss_out[0] = 0.01f * (l + logf(16.0f));
        unsigned off = 0;
        for (int e = 0; e < Esz; ++e) { offsets[e] = off; off += counts[e]; }
    }
}

// ---------------- 5: scatter tokens into expert-sorted order ----------------
__global__ void k_scatter(const int* __restrict__ top_i, const float* __restrict__ top_w,
                          const unsigned* __restrict__ offsets, unsigned* __restrict__ cursors,
                          int* __restrict__ routed_b, float* __restrict__ routed_w) {
    int b = blockIdx.x * blockDim.x + threadIdx.x;
    if (b >= Bsz) return;
#pragma unroll
    for (int k = 0; k < TOPK; ++k) {
        int e = top_i[b * 2 + k];
        unsigned p = atomicAdd(&cursors[e], 1u);
        unsigned slot = offsets[e] + p;
        routed_b[slot] = b;
        routed_w[slot] = top_w[b * 2 + k];
    }
}

// ---------------- 6: grouped bf16 WMMA GEMM per expert ----------------
// Tile: 128 tokens x 128 outputs x K=32 per iter, double-buffered LDS filled
// by GLOBAL_LOAD_ASYNC_TO_LDS_B128 (ASYNCcnt) overlapping the WMMAs.
// 8 waves -> 4x2 wave grid, each wave 32x64 = 2x4 v_wmma_f32_16x16x32_bf16.
__global__ __launch_bounds__(256)
void k_moe_gemm(const unsigned short* __restrict__ featB,   // [B][D] bf16
                const unsigned short* __restrict__ WeT,     // [E][O][D] bf16
                const float* __restrict__ b_e,              // [E][O]
                const int* __restrict__ routed_b, const float* __restrict__ routed_w,
                const unsigned* __restrict__ counts, const unsigned* __restrict__ offsets,
                float* __restrict__ out) {
    constexpr int LDSS = 40;                 // padded row stride (ushorts)
    __shared__ __align__(16) unsigned short sA[2][128 * LDSS];
    __shared__ __align__(16) unsigned short sB[2][128 * LDSS];
    __shared__ int   sTok[128];
    __shared__ float sW[128];

    int e = blockIdx.z;
    unsigned cnt = counts[e];
    unsigned m0  = blockIdx.y * 128;
    if (m0 >= cnt) return;                   // uniform: EXEC stays all-ones
    unsigned seg = offsets[e];
    int n0 = blockIdx.x * 128;

    int tid = threadIdx.x;
    if (tid < 128) {
        unsigned pos = m0 + tid;
        bool ok = pos < cnt;
        sTok[tid] = ok ? routed_b[seg + pos] : 0;
        sW[tid]   = ok ? routed_w[seg + pos] : 0.0f;
    }
    __syncthreads();

    // stage mapping: 2 threads per tile row, 16 ushorts (32B = 2 x b128) each
    int lr = tid >> 1;
    int lc = (tid & 1) * 16;
    const unsigned short* gA = featB + (size_t)sTok[lr] * Dsz + lc;
    const unsigned short* gB = WeT + ((size_t)e * Osz + n0 + lr) * Dsz + lc;
    unsigned la[2], lb[2];
#pragma unroll
    for (int b = 0; b < 2; ++b) {
        la[b] = (unsigned)(uintptr_t)&sA[b][lr * LDSS + lc];
        lb[b] = (unsigned)(uintptr_t)&sB[b][lr * LDSS + lc];
    }

    int wid  = tid >> 5, lane = tid & 31;
    int wm   = (wid & 3) * 32;               // wave row block
    int wn   = (wid >> 2) * 64;              // wave col block
    int ln16 = lane & 15, lhi = lane >> 4;

    v8f acc[2][4];
#pragma unroll
    for (int mi = 0; mi < 2; ++mi)
#pragma unroll
        for (int ni = 0; ni < 4; ++ni) acc[mi][ni] = {};

    // prologue: stage K-tile 0 into buffer 0
    async_ld_b128(la[0],      gA);
    async_ld_b128(la[0] + 16, gA + 8);
    async_ld_b128(lb[0],      gB);
    async_ld_b128(lb[0] + 16, gB + 8);
    wait_async0();
    __syncthreads();

    int cur = 0;
    for (int kt = 0; kt < Dsz; kt += 32) {
        int nxt = cur ^ 1;
        if (kt + 32 < Dsz) {                 // stage next tile while computing
            async_ld_b128(la[nxt],      gA + kt + 32);
            async_ld_b128(la[nxt] + 16, gA + kt + 40);
            async_ld_b128(lb[nxt],      gB + kt + 32);
            async_ld_b128(lb[nxt] + 16, gB + kt + 40);
        }
        if (kt + 64 < Dsz) {                 // -> global_prefetch_b8
            __builtin_prefetch((const void*)(gA + kt + 64), 0, 3);
            __builtin_prefetch((const void*)(gB + kt + 64), 0, 3);
        }

        const unsigned short* cA = sA[cur];
        const unsigned short* cB = sB[cur];
        // A frags: 16x32 bf16; lanes 0-15 hold K0-7|K16-23, lanes 16-31 K8-15|K24-31
        v16bf afrag[2], bfrag[4];
#pragma unroll
        for (int mi = 0; mi < 2; ++mi) {
            int row = wm + mi * 16 + ln16;
            int kb  = lhi * 8;
            uint4 lo = *(const uint4*)&cA[row * LDSS + kb];
            uint4 hi = *(const uint4*)&cA[row * LDSS + 16 + kb];
            afrag[mi] = make_frag(lo, hi);
        }
        // B frags: 32x16 bf16; lanes 0-15 hold K0-7|K16-23, lanes 16-31 K8-15|K24-31 (N = lane&15)
#pragma unroll
        for (int ni = 0; ni < 4; ++ni) {
            int n  = wn + ni * 16 + ln16;
            int kb = lhi * 8;
            uint4 lo = *(const uint4*)&cB[n * LDSS + kb];
            uint4 hi = *(const uint4*)&cB[n * LDSS + kb + 16];
            bfrag[ni] = make_frag(lo, hi);
        }
#pragma unroll
        for (int mi = 0; mi < 2; ++mi)
#pragma unroll
            for (int ni = 0; ni < 4; ++ni)
                acc[mi][ni] = __builtin_amdgcn_wmma_f32_16x16x32_bf16(
                    false, afrag[mi], false, bfrag[ni],
                    (short)0, acc[mi][ni], false, false);

        wait_async0();                       // my wave's async copies into nxt done
        __syncthreads();                     // everyone's done; cur free for reuse
        cur = nxt;
    }

    // epilogue: add b_e, scale by combine weight, atomic scatter-add by token
#pragma unroll
    for (int mi = 0; mi < 2; ++mi) {
#pragma unroll
        for (int ni = 0; ni < 4; ++ni) {
            int colg  = n0 + wn + ni * 16 + ln16;
            float bias = b_e[e * Osz + colg];
#pragma unroll
            for (int r = 0; r < 8; ++r) {
                int tr = wm + mi * 16 + 8 * lhi + r;  // C/D layout: VGPR r -> M=r(+8)
                unsigned pos = m0 + (unsigned)tr;
                if (pos < cnt) {
                    int tok = sTok[tr];
                    float w = sW[tr];
                    atomicAdd(&out[(size_t)tok * Osz + colg],
                              w * (acc[mi][ni][r] + bias));
                }
            }
        }
    }
}

// ---------------- host-side launcher ----------------
extern "C" void kernel_launch(void* const* d_in, const int* in_sizes, int n_in,
                              void* d_out, int out_size, void* d_ws, size_t ws_size,
                              hipStream_t stream) {
    const float* feat = (const float*)d_in[0];   // [B,D]
    const float* W_r  = (const float*)d_in[1];   // [D,E]
    const float* b_r  = (const float*)d_in[2];   // [E]
    const float* W_e  = (const float*)d_in[3];   // [E,D,O]
    const float* b_e  = (const float*)d_in[4];   // [E,O]
    float* out = (float*)d_out;                  // [B*O] logits + [1] loss

    // workspace layout
    char* w = (char*)d_ws;
    float*    pi_sum    = (float*)(w);                       // 16
    unsigned* counts    = (unsigned*)(w + 64);               // 16
    unsigned* cursors   = (unsigned*)(w + 128);              // 16
    unsigned* offsets   = (unsigned*)(w + 192);              // 16
    int*      top_i     = (int*)(w + 256);                   // B*2
    float*    top_w     = (float*)(w + 256 + 65536);         // B*2
    int*      routed_b  = (int*)(w + 256 + 2 * 65536);       // B*2
    float*    routed_w  = (float*)(w + 256 + 3 * 65536);     // B*2
    float*    blockProb = (float*)(w + 256 + 4 * 65536);     // 1024*16
    unsigned short* featB = (unsigned short*)(w + 327936);            // B*D bf16 (64MB)
    unsigned short* WeT   = (unsigned short*)(w + 327936 + 67108864); // E*O*D bf16 (64MB)

    const int nOut = Bsz * Osz + 1;

    k_init<<<(nOut + 255) / 256, 256, 0, stream>>>(out, nOut, pi_sum, counts, cursors);

    k_convert_rows<<<(Bsz * Dsz / 8 + 255) / 256, 256, 0, stream>>>(
        feat, featB, Bsz * Dsz / 8);

    k_convert_weT<<<(Esz * Osz * (Dsz / 4) + 255) / 256, 256, 0, stream>>>(W_e, WeT);

    k_router<<<Bsz / 8, 256, 0, stream>>>(feat, W_r, b_r, top_i, top_w, counts, blockProb);

    k_finalize<<<1, 32, 0, stream>>>(counts, offsets, blockProb, Bsz / 8, out + Bsz * Osz);

    k_scatter<<<(Bsz + 255) / 256, 256, 0, stream>>>(top_i, top_w, offsets, cursors,
                                                     routed_b, routed_w);

    dim3 grid(Osz / 128, Bsz / 128, Esz);   // (4 N-tiles, 64 max M-tiles, 16 experts)
    k_moe_gemm<<<grid, 256, 0, stream>>>(featB, WeT, b_e, routed_b, routed_w,
                                         counts, offsets, out);
}
